// PHGATLayer_69870527971893
// MI455X (gfx1250) — compile-verified, matrix operand order checked by reading, not run on hardware
//
#include <hip/hip_runtime.h>
#include <hip/hip_bf16.h>
#include <math.h>

// ---------------------------------------------------------------------------
// PHGAT layer for MI455X (gfx1250, wave32).
//  - 10 dense [Nx128]@[128x128]^T GEMMs via v_wmma_f32_16x16x32_bf16,
//    W staged in LDS as bf16; B fragments for all 4 K-blocks preloaded so
//    ds_load latency overlaps the WMMA chain; per-row inverse norms fused
//    into the epilogue (butterfly over the 16 lanes of each half-wave).
//  - 6 edge phases: wave-per-edge (32 lanes x float4 = one 128-row),
//    cosine sim via cross-lane butterfly, f32 atomic scatter-add (L2 atomics).
//  - Size-1 softmaxes in the reference collapse to constants; only the
//    2-way softmax over {wn,vd} mean-attention survives into the epilogue.
// ---------------------------------------------------------------------------

typedef __attribute__((ext_vector_type(16))) __bf16 v16bf;
typedef __attribute__((ext_vector_type(8)))  float  v8f;
typedef __attribute__((ext_vector_type(8)))  __bf16 bf16x8;

#define DD 128

__device__ __forceinline__ float wave_sum32(float v) {
#pragma unroll
  for (int off = 16; off > 0; off >>= 1) v += __shfl_xor(v, off, 32);
  return v;
}

// ---- out[N x 128] = X[N x 128] @ W^T (+bias); invn[r] = 1/max(||out_r||,eps)
__global__ __launch_bounds__(256)
void gemm128_wmma_kernel(const float* __restrict__ X, const float* __restrict__ W,
                         const float* __restrict__ bias, float* __restrict__ out,
                         float* __restrict__ invn, int N)
{
  __shared__ __bf16 wlds[DD * DD];                 // W row-major: wlds[j*128+k]
  for (int i = threadIdx.x; i < DD * DD; i += 256) wlds[i] = (__bf16)W[i];
  __syncthreads();

  const int wave = threadIdx.x >> 5;
  const int lane = threadIdx.x & 31;
  const int m    = lane & 15;
  const int kb   = (lane >> 4) << 3;               // 0 (lanes 0-15) or 8 (16-31)
  const int rowBase = blockIdx.x * 128 + wave * 16;

  // A fragments: 16x32 bf16 per K-block (ISA 7.12.2 16-bit A layout).
  int row = rowBase + m;
  int rc  = row < N ? row : (N - 1);               // clamp; stores are guarded
  const float4* Xr = reinterpret_cast<const float4*>(X + (size_t)rc * DD);
  v16bf afrag[4];
#pragma unroll
  for (int kblk = 0; kblk < 4; ++kblk) {
    int b0 = (kblk * 32 + kb) >> 2;                // float4 index: K..K+7
    float4 p0 = Xr[b0], p1 = Xr[b0 + 1];           // K+0..7
    float4 p2 = Xr[b0 + 4], p3 = Xr[b0 + 5];       // K+16..23
    v16bf a;
    a[0]=(__bf16)p0.x; a[1]=(__bf16)p0.y; a[2]=(__bf16)p0.z; a[3]=(__bf16)p0.w;
    a[4]=(__bf16)p1.x; a[5]=(__bf16)p1.y; a[6]=(__bf16)p1.z; a[7]=(__bf16)p1.w;
    a[8]=(__bf16)p2.x; a[9]=(__bf16)p2.y; a[10]=(__bf16)p2.z; a[11]=(__bf16)p2.w;
    a[12]=(__bf16)p3.x; a[13]=(__bf16)p3.y; a[14]=(__bf16)p3.z; a[15]=(__bf16)p3.w;
    afrag[kblk] = a;
  }

  const int n = lane & 15;
  const int rstore = rowBase + ((lane >> 4) << 3); // f32 C/D: VGPR r -> M=r or r+8
  const bool full = (rowBase + 16) <= N;           // wave-uniform fast path
  float rs[8] = {0, 0, 0, 0, 0, 0, 0, 0};         // per-row sum of squares (partial)

#pragma unroll 1
  for (int ct = 0; ct < 8; ++ct) {
    const int col = ct * 16 + n;

    // Preload B fragments for all 4 K-blocks so the ds_load latency is
    // hidden under the WMMA chain (single dscnt wait instead of four).
    // B[k][n] = W[n][k]; 16-bit B layout mirrors A (K halves per half-wave).
    v16bf bfr[4];
#pragma unroll
    for (int kblk = 0; kblk < 4; ++kblk) {
      const int base = (col << 7) + kblk * 32 + kb;
      bf16x8 q0 = *(const bf16x8*)(wlds + base);        // K..K+7
      bf16x8 q1 = *(const bf16x8*)(wlds + base + 16);   // K+16..K+23
      v16bf b;
#pragma unroll
      for (int i = 0; i < 8; ++i) { b[i] = q0[i]; b[8 + i] = q1[i]; }
      bfr[kblk] = b;
    }

    v8f acc = {};
#pragma unroll
    for (int kblk = 0; kblk < 4; ++kblk)
      acc = __builtin_amdgcn_wmma_f32_16x16x32_bf16(false, afrag[kblk], false, bfr[kblk],
                                                    (short)0, acc, false, false);

    const float bi = bias[col];                     // always valid (zero vec if none)
    float* op = out + (size_t)rstore * DD + col;
    if (full) {
#pragma unroll
      for (int r = 0; r < 8; ++r) {
        const float v = acc[r] + bi;
        rs[r] += v * v;
        op[(size_t)r * DD] = v;
      }
    } else {
#pragma unroll
      for (int r = 0; r < 8; ++r) {
        const float v = acc[r] + bi;
        rs[r] += v * v;
        if (rstore + r < N) op[(size_t)r * DD] = v;
      }
    }
  }

  // Reduce row sums over the 16 lanes of each half-wave (cols), write inv norms.
#pragma unroll
  for (int r = 0; r < 8; ++r) {
#pragma unroll
    for (int off = 8; off > 0; off >>= 1) rs[r] += __shfl_xor(rs[r], off, 32);
  }
  if ((lane & 15) == 0) {
#pragma unroll
    for (int r = 0; r < 8; ++r) {
      const int rr = rstore + r;
      if (rr < N) invn[rr] = 1.0f / fmaxf(sqrtf(rs[r]), 1e-8f);
    }
  }
}

// ---- wave-per-edge: cosine sim + scatter-add ------------------------------
template <bool SAME_MSG>
__global__ __launch_bounds__(256)
void edge_kernel(const float* __restrict__ hsim, const float* __restrict__ invs,
                 const float* __restrict__ hmsg,
                 const float* __restrict__ hdst, const float* __restrict__ invd,
                 const int* __restrict__ src, const int* __restrict__ dst,
                 int E, float scale,
                 float* __restrict__ hacc, float* __restrict__ ssum,
                 float* __restrict__ dcnt)
{
  const int gid  = blockIdx.x * blockDim.x + threadIdx.x;
  const int e    = gid >> 5;
  const int lane = threadIdx.x & 31;
  if (e >= E) return;
  const int s = src[e], d = dst[e];
  float4 av = reinterpret_cast<const float4*>(hsim + (size_t)s * DD)[lane];
  float4 bv = reinterpret_cast<const float4*>(hdst + (size_t)d * DD)[lane];
  float dot = wave_sum32(av.x * bv.x + av.y * bv.y + av.z * bv.z + av.w * bv.w);
  const float sim = dot * invs[s] * invd[d] * scale;      // uniform over wave
  float4 mv = av;
  if (!SAME_MSG) mv = reinterpret_cast<const float4*>(hmsg + (size_t)s * DD)[lane];
  float* o = hacc + (size_t)d * DD + lane * 4;
  atomicAdd(o + 0, sim * mv.x);
  atomicAdd(o + 1, sim * mv.y);
  atomicAdd(o + 2, sim * mv.z);
  atomicAdd(o + 3, sim * mv.w);
  if (lane == 0) { atomicAdd(&ssum[d], sim); atomicAdd(&dcnt[d], 1.0f); }
}

// ---- epilogues ------------------------------------------------------------
__global__ void finish_vul_kernel(const float* __restrict__ ht,
                                  const float* __restrict__ ph0, const float* __restrict__ ph1,
                                  const float* __restrict__ nh,
                                  const float* __restrict__ s0, const float* __restrict__ g0,
                                  const float* __restrict__ s1, const float* __restrict__ g1,
                                  float* __restrict__ out, int n)
{
  const int gid = blockIdx.x * blockDim.x + threadIdx.x;
  const int v = gid >> 5, q = gid & 31;
  if (v >= n) return;
  const float m0 = s0[v] / fmaxf(g0[v], 1.0f);
  const float m1 = s1[v] / fmaxf(g1[v], 1.0f);
  const float mx = fmaxf(m0, m1);
  const float e0 = expf(m0 - mx), e1 = expf(m1 - mx);
  const float inv = 1.0f / (e0 + e1);
  const float pa0 = 0.6f * e0 * inv, pa1 = 0.6f * e1 * inv;  // PRATE=0.6, NRATE=0.4
  const size_t ib = (size_t)v * DD + q * 4;
  float4 a = *reinterpret_cast<const float4*>(ph0 + ib);
  float4 b = *reinterpret_cast<const float4*>(ph1 + ib);
  float4 c = *reinterpret_cast<const float4*>(nh + ib);
  float4 t = *reinterpret_cast<const float4*>(ht + ib);
  float4 h;
  h.x = pa0 * a.x + pa1 * b.x + 0.4f * c.x;
  h.y = pa0 * a.y + pa1 * b.y + 0.4f * c.y;
  h.z = pa0 * a.z + pa1 * b.z + 0.4f * c.z;
  h.w = pa0 * a.w + pa1 * b.w + 0.4f * c.w;
  *reinterpret_cast<float4*>(out + (size_t)v * 256 + q * 4)       = t;
  *reinterpret_cast<float4*>(out + (size_t)v * 256 + 128 + q * 4) = h;
}

__global__ void finish_ent_kernel(const float* __restrict__ ht, const float* __restrict__ hr,
                                  float* __restrict__ out, int n)
{
  const int gid = blockIdx.x * blockDim.x + threadIdx.x;
  const int v = gid >> 5, q = gid & 31;
  if (v >= n) return;
  const size_t ib = (size_t)v * DD + q * 4;
  *reinterpret_cast<float4*>(out + (size_t)v * 256 + q * 4) =
      *reinterpret_cast<const float4*>(ht + ib);
  *reinterpret_cast<float4*>(out + (size_t)v * 256 + 128 + q * 4) =
      *reinterpret_cast<const float4*>(hr + ib);
}

__global__ void zero_kernel(float* __restrict__ p, size_t n)
{
  size_t i = (size_t)blockIdx.x * blockDim.x + threadIdx.x;
  const size_t stride = (size_t)gridDim.x * blockDim.x;
  for (; i < n; i += stride) p[i] = 0.0f;
}

// ---------------------------------------------------------------------------
extern "C" void kernel_launch(void* const* d_in, const int* in_sizes, int n_in,
                              void* d_out, int out_size, void* d_ws, size_t ws_size,
                              hipStream_t stream)
{
  (void)n_in; (void)out_size; (void)ws_size;
  const float* feat_vul = (const float*)d_in[0];
  const float* feat_wn  = (const float*)d_in[1];
  const float* feat_vd  = (const float*)d_in[2];
  const float* feat_dc  = (const float*)d_in[3];
  const float* We_wn2vul = (const float*)d_in[4];
  const float* We_vd2vul = (const float*)d_in[5];
  const float* We_dc2vul = (const float*)d_in[6];
  const float* We_vul2wn = (const float*)d_in[7];
  const float* We_vul2vd = (const float*)d_in[8];
  const float* We_vul2dc = (const float*)d_in[9];
  const float* Wn_vul = (const float*)d_in[10];  const float* bn_vul = (const float*)d_in[11];
  const float* Wn_wn  = (const float*)d_in[12];  const float* bn_wn  = (const float*)d_in[13];
  const float* Wn_vd  = (const float*)d_in[14];  const float* bn_vd  = (const float*)d_in[15];
  const float* Wn_dc  = (const float*)d_in[16];  const float* bn_dc  = (const float*)d_in[17];
  const int* src_wn2vul = (const int*)d_in[18];  const int* dst_wn2vul = (const int*)d_in[19];
  const int* src_vd2vul = (const int*)d_in[20];  const int* dst_vd2vul = (const int*)d_in[21];
  const int* src_dc2vul = (const int*)d_in[22];  const int* dst_dc2vul = (const int*)d_in[23];
  const int* src_vul2wn = (const int*)d_in[24];  const int* dst_vul2wn = (const int*)d_in[25];
  const int* src_vul2vd = (const int*)d_in[26];  const int* dst_vul2vd = (const int*)d_in[27];
  const int* src_vul2dc = (const int*)d_in[28];  const int* dst_vul2dc = (const int*)d_in[29];

  const int NV = in_sizes[0] / DD, NW = in_sizes[1] / DD;
  const int NU = in_sizes[2] / DD, NC = in_sizes[3] / DD;
  const int E1 = in_sizes[18], E2 = in_sizes[20], E3 = in_sizes[22];
  const int E4 = in_sizes[24], E5 = in_sizes[26], E6 = in_sizes[28];

  // ---- workspace layout ----
  float* ws = (float*)d_ws;
  size_t o = 0;
  auto take = [&](size_t nf) { float* p = ws + o; o += nf; return p; };
  float* ht_vul    = take((size_t)NV * DD);
  float* ht_wn     = take((size_t)NW * DD);
  float* ht_vd     = take((size_t)NU * DD);
  float* ht_dc     = take((size_t)NC * DD);
  float* hr_wn     = take((size_t)NW * DD);
  float* hr_vd     = take((size_t)NU * DD);
  float* hr_dc     = take((size_t)NC * DD);
  float* hr_vul_wn = take((size_t)NV * DD);
  float* hr_vul_vd = take((size_t)NV * DD);
  float* hr_vul    = take((size_t)NV * DD);
  float* in_ht_vul = take(NV); float* in_ht_wn = take(NW);
  float* in_ht_vd  = take(NU); float* in_ht_dc = take(NC);
  float* in_hr_wn  = take(NW); float* in_hr_vd = take(NU); float* in_hr_dc = take(NC);
  float* in_hv_wn  = take(NV); float* in_hv_vd = take(NV); float* in_hv    = take(NV);
  float* accBase = ws + o;                         // zeroed per call (before GEMMs)
  float* zbias  = take(DD);                        // shared all-zero bias vector
  float* ph_wn  = take((size_t)NV * DD);
  float* ph_vd  = take((size_t)NV * DD);
  float* nh     = take((size_t)NV * DD);
  float* h_wn_r = take((size_t)NW * DD);
  float* h_vd_r = take((size_t)NU * DD);
  float* h_dc_r = take((size_t)NC * DD);
  float* s_wn = take(NV); float* g_wn = take(NV);
  float* s_vd = take(NV); float* g_vd = take(NV);
  float* s_dc = take(NV); float* g_dc = take(NV);
  float* s_w  = take(NW); float* g_w  = take(NW);
  float* s_v  = take(NU); float* g_v  = take(NU);
  float* s_d  = take(NC); float* g_d  = take(NC);
  const size_t nzero = (size_t)(ws + o - accBase);

  float* out_vul = (float*)d_out;
  float* out_wn  = out_vul + (size_t)NV * 256;
  float* out_vd  = out_wn  + (size_t)NW * 256;
  float* out_dc  = out_vd  + (size_t)NU * 256;

  auto gblk = [](int N) { return dim3((unsigned)((N + 127) / 128)); };   // gemm: 128 rows/block
  auto wgrd = [](long long n) { return dim3((unsigned)((n * 32 + 255) / 256)); }; // wave-per-item

  // ---- 1) zero accumulators + zero-bias (graph-replay safe) ----
  zero_kernel<<<dim3(4096), 256, 0, stream>>>(accBase, nzero);

  // ---- 2) dense projections (bf16 WMMA) with fused row-norms ----
  gemm128_wmma_kernel<<<gblk(NV), 256, 0, stream>>>(feat_vul, Wn_vul, bn_vul, ht_vul, in_ht_vul, NV);
  gemm128_wmma_kernel<<<gblk(NW), 256, 0, stream>>>(feat_wn,  Wn_wn,  bn_wn,  ht_wn,  in_ht_wn,  NW);
  gemm128_wmma_kernel<<<gblk(NU), 256, 0, stream>>>(feat_vd,  Wn_vd,  bn_vd,  ht_vd,  in_ht_vd,  NU);
  gemm128_wmma_kernel<<<gblk(NC), 256, 0, stream>>>(feat_dc,  Wn_dc,  bn_dc,  ht_dc,  in_ht_dc,  NC);
  gemm128_wmma_kernel<<<gblk(NW), 256, 0, stream>>>(feat_wn,  We_wn2vul, zbias, hr_wn, in_hr_wn, NW);
  gemm128_wmma_kernel<<<gblk(NU), 256, 0, stream>>>(feat_vd,  We_vd2vul, zbias, hr_vd, in_hr_vd, NU);
  gemm128_wmma_kernel<<<gblk(NC), 256, 0, stream>>>(feat_dc,  We_dc2vul, zbias, hr_dc, in_hr_dc, NC);
  gemm128_wmma_kernel<<<gblk(NV), 256, 0, stream>>>(feat_vul, We_vul2wn, zbias, hr_vul_wn, in_hv_wn, NV);
  gemm128_wmma_kernel<<<gblk(NV), 256, 0, stream>>>(feat_vul, We_vul2vd, zbias, hr_vul_vd, in_hv_vd, NV);
  gemm128_wmma_kernel<<<gblk(NV), 256, 0, stream>>>(feat_vul, We_vul2dc, zbias, hr_vul,    in_hv,    NV);

  // ---- 3) edge scatter phases ----
  edge_kernel<true ><<<wgrd(E1), 256, 0, stream>>>(hr_wn, in_hr_wn, hr_wn, ht_vul, in_ht_vul,
      src_wn2vul, dst_wn2vul, E1, 1.0f, ph_wn, s_wn, g_wn);
  edge_kernel<true ><<<wgrd(E2), 256, 0, stream>>>(hr_vd, in_hr_vd, hr_vd, ht_vul, in_ht_vul,
      src_vd2vul, dst_vd2vul, E2, 1.0f, ph_vd, s_vd, g_vd);
  edge_kernel<true ><<<wgrd(E3), 256, 0, stream>>>(hr_dc, in_hr_dc, hr_dc, ht_vul, in_ht_vul,
      src_dc2vul, dst_dc2vul, E3, 0.2f /*ADJ_SIM*/, nh, s_dc, g_dc);
  // reference quirk: vul->X messages always carry hr_vul (vul2dc projection)
  edge_kernel<false><<<wgrd(E4), 256, 0, stream>>>(hr_vul_wn, in_hv_wn, hr_vul, ht_wn, in_ht_wn,
      src_vul2wn, dst_vul2wn, E4, 1.0f, h_wn_r, s_w, g_w);
  edge_kernel<false><<<wgrd(E5), 256, 0, stream>>>(hr_vul_vd, in_hv_vd, hr_vul, ht_vd, in_ht_vd,
      src_vul2vd, dst_vul2vd, E5, 1.0f, h_vd_r, s_v, g_v);
  edge_kernel<true ><<<wgrd(E6), 256, 0, stream>>>(hr_vul, in_hv, hr_vul, ht_dc, in_ht_dc,
      src_vul2dc, dst_vul2dc, E6, 1.0f, h_dc_r, s_d, g_d);

  // ---- 4) epilogues ----
  finish_vul_kernel<<<wgrd(NV), 256, 0, stream>>>(ht_vul, ph_wn, ph_vd, nh,
                                                  s_wn, g_wn, s_vd, g_vd, out_vul, NV);
  finish_ent_kernel<<<wgrd(NW), 256, 0, stream>>>(ht_wn, h_wn_r, out_wn, NW);
  finish_ent_kernel<<<wgrd(NU), 256, 0, stream>>>(ht_vd, h_vd_r, out_vd, NU);
  finish_ent_kernel<<<wgrd(NC), 256, 0, stream>>>(ht_dc, h_dc_r, out_dc, NC);
}
